// MultiLayerConstrainMultiHeadAttention_13709535609370
// MI455X (gfx1250) — matrix operand order repaired and checked by reading, not compiled, not run on hardware
//
#include <hip/hip_runtime.h>
#include <hip/hip_bf16.h>
#include <math.h>
#include <stdint.h>

// ---------------------------------------------------------------------------
// MultiLayerConstrainMultiHeadAttention for MI455X (gfx1250), bf16 WMMA path.
// B=8, S=1024, D=768, H=12, DK=64, L=3.
// ---------------------------------------------------------------------------

typedef __bf16 bf16_t;
typedef __attribute__((ext_vector_type(16))) __bf16 v16bf;
typedef __attribute__((ext_vector_type(8)))  float  v8f;

#define SEQ   1024
#define DIM   768
#define NH    12
#define DKH   64
#define NB    8
#define NL    3

#define USE_ASYNC 1   // GLOBAL_LOAD_ASYNC_TO_LDS_B128 staging (verified to assemble)

#define WMMA_BF16(a, b, c) \
    __builtin_amdgcn_wmma_f32_16x16x32_bf16(false, (a), false, (b), (short)0, (c), false, false)

union ABTile { v16bf v; float4 f4[2]; };

// A-matrix 16x32 bf16 tile (CDNA5 layout, ISA 7.12.2):
// lanes 0-15: row M=lane, K in {0-7 (V0-3), 16-23 (V4-7)}
// lanes 16-31: row M=lane-16, K in {8-15, 24-31}
__device__ inline v16bf load_A_tile(const bf16_t* __restrict__ base, int ld,
                                    int m0, int k0, int lane) {
    int row = m0 + (lane & 15);
    int hs  = (lane >> 4) * 8;
    const bf16_t* p = base + (size_t)row * ld + k0 + hs;
    ABTile t;
    t.f4[0] = *(const float4*)(p);        // K = k0+hs   .. +7
    t.f4[1] = *(const float4*)(p + 16);   // K = k0+16+hs .. +7
    return t.v;
}

// B-matrix 32x16 bf16 tile (KxN) from LDS/global row-major [n][k] with pitch.
// lanes 0-15: N=n0+lane, K={0-7, 16-23}; lanes 16-31: same N, K={8-15, 24-31}.
__device__ inline v16bf load_B_lds(const bf16_t* base, int pitch, int n0, int lane) {
    const bf16_t* p = base + (size_t)(n0 + (lane & 15)) * pitch + ((lane >> 4) * 8);
    ABTile t;
    t.f4[0] = *(const float4*)(p);
    t.f4[1] = *(const float4*)(p + 16);
    return t.v;
}

__device__ inline float hmax16(float x) {
    x = fmaxf(x, __shfl_xor(x, 1));
    x = fmaxf(x, __shfl_xor(x, 2));
    x = fmaxf(x, __shfl_xor(x, 4));
    x = fmaxf(x, __shfl_xor(x, 8));
    return x;
}
__device__ inline float hsum16(float x) {
    x += __shfl_xor(x, 1);
    x += __shfl_xor(x, 2);
    x += __shfl_xor(x, 4);
    x += __shfl_xor(x, 8);
    return x;
}

#if USE_ASYNC
#define ASYNC_COPY_B128(ldsoff, gptr) \
    asm volatile("global_load_async_to_lds_b128 %0, %1, off" :: "v"(ldsoff), "v"(gptr) : "memory")
#define ASYNC_WAIT() asm volatile("s_wait_asynccnt 0" ::: "memory")
#endif

// ---------------------------------------------------------------------------
__global__ void cvt_f32_bf16(const float* __restrict__ in, bf16_t* __restrict__ out, int n) {
    for (int i = blockIdx.x * blockDim.x + threadIdx.x; i < n; i += gridDim.x * blockDim.x)
        out[i] = (bf16_t)in[i];
}

// ---------------------------------------------------------------------------
// out[M x 768] = A[M x 768] @ W^T + bias        (W is [768 x 768] row-major [n][k])
// Block: 256 thr = 8 waves, block tile 256(M) x 64(N); wave tile 32x64.
// B tile staged once per block per k-step into LDS (double-buffered async copies).
// mode 0: bf16 row-major; mode 1: bf16 transposed-per-head (V); mode 2: f32+residual.
__global__ __launch_bounds__(256) void gemm_wmma(
    const bf16_t* __restrict__ A,
    const bf16_t* __restrict__ W,
    const float*  __restrict__ bias,
    bf16_t* __restrict__ out_bf,
    float*  __restrict__ out_f32,
    const float* __restrict__ resid,
    int mode)
{
    int lane = threadIdx.x & 31;
    int wave = threadIdx.x >> 5;
    int m0 = blockIdx.x * 256 + wave * 32;
    int n0 = blockIdx.y * 64;

    __shared__ __align__(16) bf16_t Bs[2][64][40];   // 10.0 KB, double-buffered

    int tn = threadIdx.x >> 2;          // 0..63
    int tk = (threadIdx.x & 3) * 8;     // 0,8,16,24
    const bf16_t* gsrc = W + (size_t)(n0 + tn) * DIM + tk;

    auto stage = [&](int buf, int k0) {
        unsigned ldsoff = (unsigned)(uintptr_t)&Bs[buf][tn][tk];
        const bf16_t* g = gsrc + k0;
        ASYNC_COPY_B128(ldsoff, g);
    };

    v8f acc[2][4] = {};

    stage(0, 0);
    ASYNC_WAIT();
    __syncthreads();

    for (int kt = 0; kt < DIM / 32; ++kt) {
        int k0  = kt * 32;
        int buf = kt & 1;
        if (kt + 1 < DIM / 32) stage(buf ^ 1, k0 + 32);

        v16bf a0 = load_A_tile(A, DIM, m0,      k0, lane);
        v16bf a1 = load_A_tile(A, DIM, m0 + 16, k0, lane);
#pragma unroll
        for (int nt = 0; nt < 4; ++nt) {
            v16bf b = load_B_lds(&Bs[buf][0][0], 40, nt * 16, lane);
            acc[0][nt] = WMMA_BF16(a0, b, acc[0][nt]);
            acc[1][nt] = WMMA_BF16(a1, b, acc[1][nt]);
        }
        ASYNC_WAIT();
        __syncthreads();
    }

    int col = lane & 15;
    int rhi = (lane >> 4) * 8;
#pragma unroll
    for (int mi = 0; mi < 2; ++mi) {
#pragma unroll
        for (int nt = 0; nt < 4; ++nt) {
            int n = n0 + nt * 16 + col;
            float bv = bias ? bias[n] : 0.0f;
#pragma unroll
            for (int r = 0; r < 8; ++r) {
                int row = m0 + mi * 16 + r + rhi;
                float v = acc[mi][nt][r] + bv;
                if (mode == 0) {
                    out_bf[(size_t)row * DIM + n] = (bf16_t)v;
                } else if (mode == 1) {
                    int b = row >> 10, s = row & (SEQ - 1);
                    int h = n >> 6,  dk = n & (DKH - 1);
                    out_bf[(((size_t)b * NH + h) * DKH + dk) * SEQ + s] = (bf16_t)v;
                } else {
                    out_f32[(size_t)row * DIM + n] = v + resid[(size_t)row * DIM + n];
                }
            }
        }
    }
}

// ---------------------------------------------------------------------------
// ctx = (softmax(q k^T / 8) * (dist + I)) @ v   per (b,h); flash-style stream.
// All 8 waves in a block share (b,h) and the key sweep, so K/V tiles are staged
// once per block per 32-key step via double-buffered async global->LDS copies.
// qp,kp: [B*S, D] bf16 (head h at cols h*64..). vT: [B*NH*DKH, S] bf16.
__global__ __launch_bounds__(256) void attn_wmma(
    const bf16_t* __restrict__ qp,
    const bf16_t* __restrict__ kp,
    const bf16_t* __restrict__ vT,
    const float*  __restrict__ dist,
    bf16_t* __restrict__ ctx)
{
    int lane = threadIdx.x & 31;
    int wave = threadIdx.x >> 5;
    int qblk = blockIdx.x & 7;            // S/128 = 8
    int bh   = blockIdx.x >> 3;
    int b = bh / NH, h = bh % NH;
    int m0 = qblk * 128 + wave * 16;      // query tile base within S

    __shared__ __align__(16) bf16_t Ks[2][32][72];       // keys x dk, 9.2 KB
    __shared__ __align__(16) bf16_t Vs[2][64][40];       // dk x keys, 10.0 KB
    __shared__ __align__(16) bf16_t pstage[8][16][40];   // per-wave P staging

    const bf16_t* qbase = qp + (size_t)b * SEQ * DIM;
    const bf16_t* kbase = kp + (size_t)b * SEQ * DIM;
    const bf16_t* vbase = vT + ((size_t)b * NH + h) * DKH * SEQ;
    const float*  dbase = dist + (size_t)b * SEQ * SEQ;

    // staging assignments (256 threads, one B128 each per tile)
    int krow = threadIdx.x >> 3, kq = (threadIdx.x & 7) * 8;   // K: 32 rows x 64
    int vrow = threadIdx.x >> 2, vq = (threadIdx.x & 3) * 8;   // V: 64 rows x 32
    auto stageKV = [&](int buf, int j0) {
        const bf16_t* gk = kbase + (size_t)(j0 + krow) * DIM + h * DKH + kq;
        unsigned lk = (unsigned)(uintptr_t)&Ks[buf][krow][kq];
        ASYNC_COPY_B128(lk, gk);
        const bf16_t* gv = vbase + (size_t)vrow * SEQ + j0 + vq;
        unsigned lv = (unsigned)(uintptr_t)&Vs[buf][vrow][vq];
        ASYNC_COPY_B128(lv, gv);
    };

    // q tile 16x64 as two K=32 A-matrices, resident for the whole sweep
    v16bf qa0 = load_A_tile(qbase, DIM, m0, h * DKH +  0, lane);
    v16bf qa1 = load_A_tile(qbase, DIM, m0, h * DKH + 32, lane);

    v8f acc[4] = {};
    float mrow[8], lrow[8];
#pragma unroll
    for (int r = 0; r < 8; ++r) { mrow[r] = -1e30f; lrow[r] = 0.0f; }

    const float scale = 0.125f;           // 1/sqrt(DKH)
    int col = lane & 15;
    int rhi = (lane >> 4) * 8;

    stageKV(0, 0);
    ASYNC_WAIT();
    __syncthreads();

    for (int jt = 0; jt < SEQ / 32; ++jt) {
        int j0  = jt * 32;
        int buf = jt & 1;
        if (jt + 1 < SEQ / 32) stageKV(buf ^ 1, j0 + 32);

        // ---- scores for two 16-key subtiles (B tiles from LDS, rows = keys)
        v8f s0 = {}, s1 = {};
        s0 = WMMA_BF16(qa0, load_B_lds(&Ks[buf][0][0],  72,  0, lane), s0);
        s0 = WMMA_BF16(qa1, load_B_lds(&Ks[buf][0][32], 72,  0, lane), s0);
        s1 = WMMA_BF16(qa0, load_B_lds(&Ks[buf][0][0],  72, 16, lane), s1);
        s1 = WMMA_BF16(qa1, load_B_lds(&Ks[buf][0][32], 72, 16, lane), s1);

        // ---- online softmax (denominator is UNMASKED sum, mask applied after)
        float f[8], p0[8], p1[8];
#pragma unroll
        for (int r = 0; r < 8; ++r) {
            float a0 = s0[r] * scale, a1 = s1[r] * scale;
            float t = hmax16(fmaxf(a0, a1));
            float mnew = fmaxf(mrow[r], t);
            float fr = __expf(mrow[r] - mnew);
            p0[r] = __expf(a0 - mnew);
            p1[r] = __expf(a1 - mnew);
            lrow[r] = lrow[r] * fr + hsum16(p0[r] + p1[r]);
            mrow[r] = mnew;
            f[r] = fr;
        }
#pragma unroll
        for (int nt = 0; nt < 4; ++nt)
#pragma unroll
            for (int r = 0; r < 8; ++r) acc[nt][r] *= f[r];

        // ---- mask-multiply and stage P (C-layout -> row-major in LDS)
#pragma unroll
        for (int r = 0; r < 8; ++r) {
            int q  = m0 + r + rhi;
            int ja = j0 + col, jb = j0 + 16 + col;
            float mk0 = dbase[(size_t)q * SEQ + ja] + (q == ja ? 1.0f : 0.0f);
            float mk1 = dbase[(size_t)q * SEQ + jb] + (q == jb ? 1.0f : 0.0f);
            pstage[wave][r + rhi][col]      = (bf16_t)(p0[r] * mk0);
            pstage[wave][r + rhi][16 + col] = (bf16_t)(p1[r] * mk1);
        }
        asm volatile("s_wait_dscnt 0" ::: "memory");  // intra-wave LDS RAW

        // ---- reload P as a 16x32 A-matrix from LDS
        v16bf pa;
        {
            int row = lane & 15;
            int hs  = (lane >> 4) * 8;
            const bf16_t* p = &pstage[wave][row][hs];
            ABTile t;
            t.f4[0] = *(const float4*)(p);
            t.f4[1] = *(const float4*)(p + 16);
            pa = t.v;
        }

        // ---- ctx += P @ V  (V^T tile in LDS: rows = dk, cols = keys)
#pragma unroll
        for (int nt = 0; nt < 4; ++nt) {
            v16bf vb = load_B_lds(&Vs[buf][0][0], 40, nt * 16, lane);
            acc[nt] = WMMA_BF16(pa, vb, acc[nt]);
        }

        ASYNC_WAIT();
        __syncthreads();
    }

    // ---- normalize and store ctx (bf16, row-major [B*S, D])
#pragma unroll
    for (int nt = 0; nt < 4; ++nt)
#pragma unroll
        for (int r = 0; r < 8; ++r) {
            int q = m0 + r + rhi;
            int n = h * DKH + nt * 16 + col;
            ctx[((size_t)b * SEQ + q) * DIM + n] = (bf16_t)(acc[nt][r] / lrow[r]);
        }
}

// ---------------------------------------------------------------------------
extern "C" void kernel_launch(void* const* d_in, const int* in_sizes, int n_in,
                              void* d_out, int out_size, void* d_ws, size_t ws_size,
                              hipStream_t stream) {
    const float* Qf   = (const float*)d_in[0];
    const float* Kf   = (const float*)d_in[1];
    const float* Vf   = (const float*)d_in[2];
    const float* dist = (const float*)d_in[3];
    const float* Wq   = (const float*)d_in[4];
    const float* bq   = (const float*)d_in[5];
    const float* Wk   = (const float*)d_in[6];
    const float* bk   = (const float*)d_in[7];
    const float* Wv   = (const float*)d_in[8];
    const float* bv   = (const float*)d_in[9];
    const float* Wo   = (const float*)d_in[10];
    const float* bo   = (const float*)d_in[11];
    float* outF = (float*)d_out;

    const size_t BSD = (size_t)NB * SEQ * DIM;   // 6,291,456
    const size_t WL  = (size_t)NL * DIM * DIM;   // 1,769,472

    bf16_t* ws = (bf16_t*)d_ws;
    bf16_t* Qb   = ws;
    bf16_t* Kb   = Qb  + BSD;
    bf16_t* Vb   = Kb  + BSD;
    bf16_t* Wqb  = Vb  + BSD;
    bf16_t* Wkb  = Wqb + WL;
    bf16_t* Wvb  = Wkb + WL;
    bf16_t* Wob  = Wvb + WL;
    bf16_t* qp   = Wob + WL;
    bf16_t* kp   = qp  + BSD;
    bf16_t* vTb  = kp  + BSD;
    bf16_t* ctx  = vTb + BSD;
    bf16_t* outb = ctx + BSD;

    // f32 -> bf16 staging
    cvt_f32_bf16<<<4096, 256, 0, stream>>>(Qf, Qb, (int)BSD);
    cvt_f32_bf16<<<4096, 256, 0, stream>>>(Kf, Kb, (int)BSD);
    cvt_f32_bf16<<<4096, 256, 0, stream>>>(Vf, Vb, (int)BSD);
    cvt_f32_bf16<<<2048, 256, 0, stream>>>(Wq, Wqb, (int)WL);
    cvt_f32_bf16<<<2048, 256, 0, stream>>>(Wk, Wkb, (int)WL);
    cvt_f32_bf16<<<2048, 256, 0, stream>>>(Wv, Wvb, (int)WL);
    cvt_f32_bf16<<<2048, 256, 0, stream>>>(Wo, Wob, (int)WL);

    dim3 ggrid(8192 / 256, DIM / 64);            // 32 x 12
    dim3 agrid(NB * NH * (SEQ / 128));           // 768

    for (int i = 0; i < NL; ++i) {
        const bf16_t* ksrc = (i == 0) ? Kb : outb;
        const bf16_t* vsrc = (i == 0) ? Vb : outb;
        gemm_wmma<<<ggrid, 256, 0, stream>>>(Qb,   Wqb + (size_t)i * DIM * DIM, bq + i * DIM,
                                             qp,  nullptr, nullptr, 0);
        gemm_wmma<<<ggrid, 256, 0, stream>>>(ksrc, Wkb + (size_t)i * DIM * DIM, bk + i * DIM,
                                             kp,  nullptr, nullptr, 0);
        gemm_wmma<<<ggrid, 256, 0, stream>>>(vsrc, Wvb + (size_t)i * DIM * DIM, bv + i * DIM,
                                             vTb, nullptr, nullptr, 1);
        attn_wmma<<<agrid, 256, 0, stream>>>(qp, kp, vTb, dist, ctx);
        if (i < NL - 1) {
            gemm_wmma<<<ggrid, 256, 0, stream>>>(ctx, Wob + (size_t)i * DIM * DIM, bo + i * DIM,
                                                 outb, nullptr, nullptr, 0);
        } else {
            gemm_wmma<<<ggrid, 256, 0, stream>>>(ctx, Wob + (size_t)i * DIM * DIM, bo + i * DIM,
                                                 nullptr, outF, Vf, 2);
        }
    }
}
